// PointUNetObj_20590073217623
// MI455X (gfx1250) — matrix operand order repaired and checked by reading, not compiled, not run on hardware
//
#include <hip/hip_runtime.h>
#include <hip/hip_bf16.h>

typedef __attribute__((ext_vector_type(16))) _Float16 v16h;
typedef __attribute__((ext_vector_type(8)))  _Float16 v8h;
typedef __attribute__((ext_vector_type(8)))  float    v8f;

#define DEVINL __device__ __forceinline__

// y layout: (B, 3, 512, 1); y_[b, m, c] = y[((b*3 + c)*512) + m]
DEVINL float yAt(const float* y, int b, int m, int c) {
  return y[((size_t)b * 3 + c) * 512 + m];
}

// Deterministic permutation of [0,512) per batch (stand-in for jax.random.permutation)
DEVINL int poolperm(int b, int j) {
  return ((j * 269) & 511) ^ ((b * 211) & 511);
}

DEVINL unsigned hashu(unsigned x) {
  x ^= x >> 16; x *= 0x7feb352dU; x ^= x >> 15; x *= 0x846ca68bU; x ^= x >> 16;
  return x;
}

// Deterministic pseudo-normal (Box-Muller) for random_walk noise
DEVINL float gauss3(int b, int p, int c) {
  unsigned h1 = hashu(((unsigned)(b * 1152 + p) << 2) ^ (unsigned)c ^ 0x9e3779b9U);
  unsigned h2 = hashu(h1 + 0x85ebca6bU);
  float u1 = ((h1 >> 8) + 1) * (1.0f / 16777217.0f);
  float u2 = (h2 >> 8) * (1.0f / 16777216.0f);
  return sqrtf(-2.0f * logf(u1)) * cosf(6.28318530718f * u2);
}

// ---------------------------------------------------------------------------
// CDNA5 v_wmma_f32_16x16x32_f16 lane layouts (wave32):
//   A 16x32: lane holds row rbase+(lane&15), halves = K = k0+hi*16+(0..15) (this code's k order)
//   B 32x16: lane holds column c0+(lane&15), halves = K = k0+hi*16+(0..15) (this code's k order, same as A)
//   C 16x16: vgpr v -> row v + 8*hi, column c0 + (lane&15)
//
// A-fragment (16 rows x KPAD) is loaded to registers ONCE per row-tile and
// reused across all column tiles. Column tiles are processed in PAIRS
// (two independent accumulators) so the WMMA->VALU RAW hazard slots
// (4 co-exec NOPs for f16 WMMA, ISA 7.12.1) are filled with real work.
// ---------------------------------------------------------------------------
template <int NCH>
struct AFrag { v16h a[NCH]; };

DEVINL v16h cat16(v8h lo, v8h hi) {
  return __builtin_shufflevector(lo, hi, 0, 1, 2, 3, 4, 5, 6, 7,
                                 8, 9, 10, 11, 12, 13, 14, 15);
}

template <int KPAD>
DEVINL AFrag<KPAD / 32> load_afrag(const _Float16* __restrict__ A, int ldA,
                                   int rbase, int lane) {
  AFrag<KPAD / 32> F;
  const int hi = lane >> 4, r = lane & 15;
  const _Float16* rowA = A + (size_t)(rbase + r) * ldA;
#pragma unroll
  for (int i = 0; i < KPAD / 32; ++i) {
    v8h a0 = *(const v8h*)(rowA + i * 32 + hi * 16);
    v8h a1 = *(const v8h*)(rowA + i * 32 + hi * 16 + 8);
    F.a[i] = cat16(a0, a1);
  }
  return F;
}

// Single column tile over a cached A-fragment
template <int KPAD>
DEVINL v8f wmma_tiles(const AFrag<KPAD / 32>& F, const _Float16* __restrict__ Wh,
                      int c0, int lane) {
  const int hi = lane >> 4, r = lane & 15;
  const _Float16* rowW = Wh + (size_t)(c0 + r) * KPAD;
  v8f acc = {};
#pragma unroll
  for (int i = 0; i < KPAD / 32; ++i) {
    __builtin_prefetch(rowW + i * 32 + 64, 0, 1);   // global_prefetch_b8
    v16h bf = cat16(*(const v8h*)(rowW + i * 32 + hi * 16),
                    *(const v8h*)(rowW + i * 32 + hi * 16 + 8));
    acc = __builtin_amdgcn_wmma_f32_16x16x32_f16(false, F.a[i], false, bf,
                                                 (short)0, acc, false, false);
  }
  return acc;
}

// Paired column tiles (two accumulators) over a cached A-fragment
template <int KPAD>
DEVINL void wmma_tiles2(const AFrag<KPAD / 32>& F, const _Float16* __restrict__ Wh,
                        int c0a, int c0b, int lane, v8f& accA, v8f& accB) {
  const int hi = lane >> 4, r = lane & 15;
  const _Float16* rowWa = Wh + (size_t)(c0a + r) * KPAD;
  const _Float16* rowWb = Wh + (size_t)(c0b + r) * KPAD;
  accA = {}; accB = {};
#pragma unroll
  for (int i = 0; i < KPAD / 32; ++i) {
    __builtin_prefetch(rowWa + i * 32 + 64, 0, 1);
    v16h bfa = cat16(*(const v8h*)(rowWa + i * 32 + hi * 16),
                     *(const v8h*)(rowWa + i * 32 + hi * 16 + 8));
    v16h bfb = cat16(*(const v8h*)(rowWb + i * 32 + hi * 16),
                     *(const v8h*)(rowWb + i * 32 + hi * 16 + 8));
    accA = __builtin_amdgcn_wmma_f32_16x16x32_f16(false, F.a[i], false, bfa,
                                                  (short)0, accA, false, false);
    accB = __builtin_amdgcn_wmma_f32_16x16x32_f16(false, F.a[i], false, bfb,
                                                  (short)0, accB, false, false);
  }
}

// Paired streaming variant (A loaded per chunk, shared by both tiles) for fc1
template <int KPAD>
DEVINL void mlp_tile2(const _Float16* __restrict__ A, int ldA,
                      const _Float16* __restrict__ Wh,
                      int rbase, int c0a, int c0b, int lane, v8f& accA, v8f& accB) {
  const int hi = lane >> 4, r = lane & 15;
  const _Float16* rowA  = A + (size_t)(rbase + r) * ldA;
  const _Float16* rowWa = Wh + (size_t)(c0a + r) * KPAD;
  const _Float16* rowWb = Wh + (size_t)(c0b + r) * KPAD;
  accA = {}; accB = {};
#pragma unroll
  for (int k0 = 0; k0 < KPAD; k0 += 32) {
    __builtin_prefetch(rowWa + k0 + 64, 0, 1);
    v16h af  = cat16(*(const v8h*)(rowA + k0 + hi * 16),
                     *(const v8h*)(rowA + k0 + hi * 16 + 8));
    v16h bfa = cat16(*(const v8h*)(rowWa + k0 + hi * 16),
                     *(const v8h*)(rowWa + k0 + hi * 16 + 8));
    v16h bfb = cat16(*(const v8h*)(rowWb + k0 + hi * 16),
                     *(const v8h*)(rowWb + k0 + hi * 16 + 8));
    accA = __builtin_amdgcn_wmma_f32_16x16x32_f16(false, af, false, bfa,
                                                  (short)0, accA, false, false);
    accB = __builtin_amdgcn_wmma_f32_16x16x32_f16(false, af, false, bfb,
                                                  (short)0, accB, false, false);
  }
}

// Epilogue: relu((acc + b)*s + t) -> f16 into LDS (next layer's A buffer)
DEVINL void epi_store_lds(v8f acc, const float* __restrict__ b, const float* __restrict__ s,
                          const float* __restrict__ t, _Float16* sOut, int ldOut,
                          int rbase, int c0, int lane) {
  const int hi = lane >> 4;
  const int col = c0 + (lane & 15);
  const float bb = b[col], ss = s[col], tt = t[col];
#pragma unroll
  for (int v = 0; v < 8; ++v) {
    float val = fmaxf((acc[v] + bb) * ss + tt, 0.0f);
    sOut[(size_t)(rbase + v + 8 * hi) * ldOut + col] = (_Float16)val;
  }
}

// Epilogue: relu(...) then max over tile rows into LDS int slots.
// ReLU outputs are >= 0 so int bit-pattern compare == float compare.
DEVINL void epi_max_lds(v8f acc, const float* b, const float* s, const float* t,
                        int* sMax, int c0, int lane) {
  const int col = c0 + (lane & 15);
  const float bb = b[col], ss = s[col], tt = t[col];
  float m = 0.0f;
#pragma unroll
  for (int v = 0; v < 8; ++v) m = fmaxf(m, fmaxf((acc[v] + bb) * ss + tt, 0.0f));
  atomicMax(&sMax[col], __float_as_int(m));
}

// Same, but into a global f32 buffer (pre-zeroed)
DEVINL void epi_max_global(v8f acc, const float* b, const float* s, const float* t,
                           float* g, int c0, int lane) {
  const int col = c0 + (lane & 15);
  const float bb = b[col], ss = s[col], tt = t[col];
  float m = 0.0f;
#pragma unroll
  for (int v = 0; v < 8; ++v) m = fmaxf(m, fmaxf((acc[v] + bb) * ss + tt, 0.0f));
  atomicMax((int*)&g[col], __float_as_int(m));
}

// ---------------------------------------------------------------------------
// Weight prep: f32 (co x K) -> f16 zero-padded (copad x KPAD)
// ---------------------------------------------------------------------------
__global__ void k_wprep(const float* __restrict__ W, _Float16* __restrict__ out,
                        int co, int K, int KPAD, int total) {
  int i = blockIdx.x * blockDim.x + threadIdx.x;
  if (i >= total) return;
  int r = i / KPAD, c = i % KPAD;
  out[i] = (r < co && c < K) ? (_Float16)W[(size_t)r * K + c] : (_Float16)0.0f;
}

__global__ void k_tohalf(const float* __restrict__ in, _Float16* __restrict__ out, int n) {
  int i = blockIdx.x * blockDim.x + threadIdx.x;
  if (i < n) out[i] = (_Float16)in[i];
}

__global__ void k_zero(float* p, int n) {
  int i = blockIdx.x * blockDim.x + threadIdx.x;
  if (i < n) p[i] = 0.0f;
}

// ---------------------------------------------------------------------------
// Stage 1: netR_1 (6 -> 64 -> 128) over x(B,6,512,64), max over k=64.
// One block per (b,n). Output xfeat (B, 512, 128) f32.
// ---------------------------------------------------------------------------
__global__ void k_net1(const float* __restrict__ x,
                       const _Float16* hW0, const float* b0, const float* s0, const float* t0,
                       const _Float16* hW1, const float* b1, const float* s1, const float* t1,
                       float* __restrict__ xfeat) {
  __shared__ __align__(16) _Float16 sIn[64 * 32];   // 64 samples x 6 ch (padded)
  __shared__ __align__(16) _Float16 sMid[64 * 64];
  __shared__ int sMax[128];
  const int blk = blockIdx.x;
  const int n = blk % 512, b = blk / 512;
  const int tid = threadIdx.x;
  for (int i = tid; i < 64 * 32; i += 128) {
    int k = i >> 5, c = i & 31;
    float v = (c < 6) ? x[(((size_t)b * 6 + c) * 512 + n) * 64 + k] : 0.0f;
    sIn[i] = (_Float16)v;
  }
  if (tid < 128) sMax[tid] = 0;
  __syncthreads();
  const int lane = tid & 31, wave = tid >> 5;
  const int rbase = wave * 16;
  {
    auto F = load_afrag<32>(sIn, 32, rbase, lane);   // layer0: K=32(pad), co=64
#pragma unroll
    for (int ct = 0; ct < 4; ct += 2) {
      v8f a0, a1;
      wmma_tiles2<32>(F, hW0, ct * 16, (ct + 1) * 16, lane, a0, a1);
      epi_store_lds(a0, b0, s0, t0, sMid, 64, rbase, ct * 16, lane);
      epi_store_lds(a1, b0, s0, t0, sMid, 64, rbase, (ct + 1) * 16, lane);
    }
  }
  __syncthreads();
  {
    auto F = load_afrag<64>(sMid, 64, rbase, lane);  // layer1: K=64, co=128, + max
#pragma unroll
    for (int ct = 0; ct < 8; ct += 2) {
      v8f a0, a1;
      wmma_tiles2<64>(F, hW1, ct * 16, (ct + 1) * 16, lane, a0, a1);
      epi_max_lds(a0, b1, s1, t1, sMax, ct * 16, lane);
      epi_max_lds(a1, b1, s1, t1, sMax, (ct + 1) * 16, lane);
    }
  }
  __syncthreads();
  if (tid < 128)
    xfeat[((size_t)b * 512 + n) * 128 + tid] = __int_as_float(sMax[tid]);
}

// ---------------------------------------------------------------------------
// KNN: 64 nearest of ncand candidates per center; radius rule -> center index.
// ---------------------------------------------------------------------------
__global__ void k_knn(const float* __restrict__ y, int ncand, int npoint, int pooled,
                      int* __restrict__ idxOut) {
  __shared__ float d2[512];
  __shared__ float sv[128];
  __shared__ int   si[128];
  const int blk = blockIdx.x;
  const int i = blk % npoint, b = blk / npoint;
  const int tid = threadIdx.x;
  const int cm = pooled ? poolperm(b, i) : i;
  const float cx = yAt(y, b, cm, 0), cy = yAt(y, b, cm, 1), cz = yAt(y, b, cm, 2);
  for (int j = tid; j < ncand; j += 128) {
    int m = pooled ? poolperm(b, j) : j;
    float dx = yAt(y, b, m, 0) - cx;
    float dy = yAt(y, b, m, 1) - cy;
    float dz = yAt(y, b, m, 2) - cz;
    d2[j] = dx * dx + dy * dy + dz * dz;
  }
  __syncthreads();
  for (int kk = 0; kk < 64; ++kk) {
    float best = 1e30f; int bj = 0;
    for (int j = tid; j < ncand; j += 128)
      if (d2[j] < best) { best = d2[j]; bj = j; }
    sv[tid] = best; si[tid] = bj;
    __syncthreads();
    for (int off = 64; off > 0; off >>= 1) {
      if (tid < off) {
        bool take = (sv[tid + off] < sv[tid]) ||
                    (sv[tid + off] == sv[tid] && si[tid + off] < si[tid]);
        if (take) { sv[tid] = sv[tid + off]; si[tid] = si[tid + off]; }
      }
      __syncthreads();
    }
    if (tid == 0) {
      int j = si[0]; float v = sv[0];
      idxOut[((size_t)b * npoint + i) * 64 + kk] = (v > 0.015f) ? i : j;
      d2[j] = 1e30f;
    }
    __syncthreads();
  }
}

// ---------------------------------------------------------------------------
// Fused group + 2-layer MLP (131(pad160) -> 128 -> 256) + max over 64 nbrs.
// netR_4: npoint=128, pooled=0.  netR_2: npoint=32, pooled=1.
// ---------------------------------------------------------------------------
__global__ void k_group_mlp(const float* __restrict__ y, const float* __restrict__ xfeat,
                            const int* __restrict__ idx, int npoint, int pooled,
                            const _Float16* hW0, const float* b0, const float* s0, const float* t0,
                            const _Float16* hW1, const float* b1, const float* s1, const float* t1,
                            float* __restrict__ outfeat) {
  __shared__ __align__(16) _Float16 sIn[64 * 160];   // 64 nbrs x 131 ch, pad 160
  __shared__ __align__(16) _Float16 sMid[64 * 128];
  __shared__ int sMax[256];
  const int blk = blockIdx.x;
  const int i = blk % npoint, b = blk / npoint;
  const int tid = threadIdx.x;
  const int cm = pooled ? poolperm(b, i) : i;
  const float cx = yAt(y, b, cm, 0), cy = yAt(y, b, cm, 1), cz = yAt(y, b, cm, 2);
  const int* myidx = idx + ((size_t)b * npoint + i) * 64;
  for (int e = tid; e < 64 * 160; e += 128) {
    int k = e / 160, c = e % 160;
    float v = 0.0f;
    if (c < 131) {
      int j = myidx[k];
      int m = pooled ? poolperm(b, j) : j;
      if (c == 0)      v = yAt(y, b, m, 0) - cx;
      else if (c == 1) v = yAt(y, b, m, 1) - cy;
      else if (c == 2) v = yAt(y, b, m, 2) - cz;
      else             v = xfeat[((size_t)b * 512 + m) * 128 + (c - 3)];
    }
    sIn[e] = (_Float16)v;
  }
  for (int c = tid; c < 256; c += 128) sMax[c] = 0;
  __syncthreads();
  const int lane = tid & 31, wave = tid >> 5;
  const int rbase = wave * 16;
  {
    auto F = load_afrag<160>(sIn, 160, rbase, lane);  // layer0: K=160(pad), co=128
#pragma unroll
    for (int ct = 0; ct < 8; ct += 2) {
      v8f a0, a1;
      wmma_tiles2<160>(F, hW0, ct * 16, (ct + 1) * 16, lane, a0, a1);
      epi_store_lds(a0, b0, s0, t0, sMid, 128, rbase, ct * 16, lane);
      epi_store_lds(a1, b0, s0, t0, sMid, 128, rbase, (ct + 1) * 16, lane);
    }
  }
  __syncthreads();
  {
    auto F = load_afrag<128>(sMid, 128, rbase, lane); // layer1: K=128, co=256, + max
#pragma unroll
    for (int ct = 0; ct < 16; ct += 2) {
      v8f a0, a1;
      wmma_tiles2<128>(F, hW1, ct * 16, (ct + 1) * 16, lane, a0, a1);
      epi_max_lds(a0, b1, s1, t1, sMax, ct * 16, lane);
      epi_max_lds(a1, b1, s1, t1, sMax, (ct + 1) * 16, lane);
    }
  }
  __syncthreads();
  for (int c = tid; c < 256; c += 128)
    outfeat[((size_t)b * npoint + i) * 256 + c] = __int_as_float(sMax[c]);
}

// ---------------------------------------------------------------------------
// Stage 3: assemble x3 (259(pad288) ch, 1152 points per batch) then
// netR_3 (-> 256 -> 512 -> 1024) with global atomic-max over points.
// One block per (b, chunk of 32 points); 36 chunks per batch.
// Row-tiles (rt) outer; col tiles paired: (ct, ct+4), ct advancing by 8.
// ---------------------------------------------------------------------------
__global__ void k_net3(const float* __restrict__ y, const float* __restrict__ h2feat,
                       const float* __restrict__ h4feat,
                       const _Float16* hW0, const float* b0, const float* s0, const float* t0,
                       const _Float16* hW1, const float* b1, const float* s1, const float* t1,
                       const _Float16* hW2, const float* b2, const float* s2, const float* t2,
                       float* __restrict__ feat) {
  __shared__ __align__(16) _Float16 sA[32 * 512];   // input (ld 512) then layer1 out
  __shared__ __align__(16) _Float16 sB[32 * 256];   // layer0 out
  const int blk = blockIdx.x;
  const int chunk = blk % 36, b = blk / 36;
  const int tid = threadIdx.x;
  for (int e = tid; e < 32 * 288; e += 128) {
    int p = e / 288, c = e % 288;
    int P = chunk * 32 + p;
    float v = 0.0f;
    if (c < 259) {
      if (P < 1024) {                     // random-walk upsampled points
        int j = P >> 5;                   // pooled center 0..31
        int m = poolperm(b, j);
        if (c < 3) v = yAt(y, b, m, c) + 0.05f * gauss3(b, P, c);
        else       v = h2feat[((size_t)b * 32 + j) * 256 + (c - 3)];
      } else {                            // x_cat points
        int q = P - 1024;                 // 0..127
        if (c < 3) v = yAt(y, b, q, c);
        else       v = h4feat[((size_t)b * 128 + q) * 256 + (c - 3)];
      }
    }
    sA[p * 512 + c] = (_Float16)v;
  }
  __syncthreads();
  const int lane = tid & 31, wave = tid >> 5;
#pragma unroll
  for (int rt = 0; rt < 2; ++rt) {        // layer0: K=288(pad), co=256
    auto F = load_afrag<288>(sA, 512, rt * 16, lane);
    for (int ct = wave; ct < 16; ct += 8) {
      v8f a0, a1;
      wmma_tiles2<288>(F, hW0, ct * 16, (ct + 4) * 16, lane, a0, a1);
      epi_store_lds(a0, b0, s0, t0, sB, 256, rt * 16, ct * 16, lane);
      epi_store_lds(a1, b0, s0, t0, sB, 256, rt * 16, (ct + 4) * 16, lane);
    }
  }
  __syncthreads();
#pragma unroll
  for (int rt = 0; rt < 2; ++rt) {        // layer1: K=256, co=512 (overwrites sA)
    auto F = load_afrag<256>(sB, 256, rt * 16, lane);
    for (int ct = wave; ct < 32; ct += 8) {
      v8f a0, a1;
      wmma_tiles2<256>(F, hW1, ct * 16, (ct + 4) * 16, lane, a0, a1);
      epi_store_lds(a0, b1, s1, t1, sA, 512, rt * 16, ct * 16, lane);
      epi_store_lds(a1, b1, s1, t1, sA, 512, rt * 16, (ct + 4) * 16, lane);
    }
  }
  __syncthreads();
#pragma unroll
  for (int rt = 0; rt < 2; ++rt) {        // layer2: K=512, co=1024, global max
    auto F = load_afrag<512>(sA, 512, rt * 16, lane);
    for (int ct = wave; ct < 64; ct += 8) {
      v8f a0, a1;
      wmma_tiles2<512>(F, hW2, ct * 16, (ct + 4) * 16, lane, a0, a1);
      epi_max_global(a0, b2, s2, t2, feat + (size_t)b * 1024, ct * 16, lane);
      epi_max_global(a1, b2, s2, t2, feat + (size_t)b * 1024, (ct + 4) * 16, lane);
    }
  }
}

// ---------------------------------------------------------------------------
// FC head: f = relu((feat @ Wf1^T + b)*s + t); out = f @ Wf2^T + b2.
// featH: (32,1024) f16 global.  Wf2 padded to 96 rows; store guarded to 87.
// ---------------------------------------------------------------------------
__global__ void k_fc(const _Float16* __restrict__ featH,
                     const _Float16* hWf1, const float* bf1, const float* sf1, const float* tf1,
                     const _Float16* hWf2, const float* bf2, float* __restrict__ out) {
  __shared__ __align__(16) _Float16 sMid[32 * 512];
  const int tid = threadIdx.x;
  const int lane = tid & 31, wave = tid >> 5;
  // fc1: M=32, K=1024, co=512. Tiles t: rbase=(t&1)*16, c0=(t>>1)*16.
  // Pair (t, t+4): same rbase, shared streaming A.
  for (int t = wave; t < 64; t += 8) {
    int rbase = (t & 1) * 16;
    v8f a0, a1;
    mlp_tile2<1024>(featH, 1024, hWf1, rbase, (t >> 1) * 16, ((t + 4) >> 1) * 16,
                    lane, a0, a1);
    epi_store_lds(a0, bf1, sf1, tf1, sMid, 512, rbase, (t >> 1) * 16, lane);
    epi_store_lds(a1, bf1, sf1, tf1, sMid, 512, rbase, ((t + 4) >> 1) * 16, lane);
  }
  __syncthreads();
#pragma unroll
  for (int rt = 0; rt < 2; ++rt) {        // fc2: M=32, K=512, co=96(pad) -> 87
    auto F = load_afrag<512>(sMid, 512, rt * 16, lane);
    for (int ct = wave; ct < 6; ct += 4) {
      v8f acc = wmma_tiles<512>(F, hWf2, ct * 16, lane);
      int hi = lane >> 4, col = ct * 16 + (lane & 15);
      if (col < 87) {
        float bb = bf2[col];
#pragma unroll
        for (int v = 0; v < 8; ++v)
          out[(size_t)(rt * 16 + v + 8 * hi) * 87 + col] = acc[v] + bb;
      }
    }
  }
}

// ---------------------------------------------------------------------------
extern "C" void kernel_launch(void* const* d_in, const int* in_sizes, int n_in,
                              void* d_out, int out_size, void* d_ws, size_t ws_size,
                              hipStream_t stream) {
  (void)in_sizes; (void)n_in; (void)out_size; (void)ws_size;
  const float* x = (const float*)d_in[0];
  const float* y = (const float*)d_in[1];
  // netR_1
  const float *W10 = (const float*)d_in[2],  *b10 = (const float*)d_in[3],
              *s10 = (const float*)d_in[4],  *t10 = (const float*)d_in[5];
  const float *W11 = (const float*)d_in[6],  *b11 = (const float*)d_in[7],
              *s11 = (const float*)d_in[8],  *t11 = (const float*)d_in[9];
  // netR_2
  const float *W20 = (const float*)d_in[10], *b20 = (const float*)d_in[11],
              *s20 = (const float*)d_in[12], *t20 = (const float*)d_in[13];
  const float *W21 = (const float*)d_in[14], *b21 = (const float*)d_in[15],
              *s21 = (const float*)d_in[16], *t21 = (const float*)d_in[17];
  // netR_4
  const float *W40 = (const float*)d_in[18], *b40 = (const float*)d_in[19],
              *s40 = (const float*)d_in[20], *t40 = (const float*)d_in[21];
  const float *W41 = (const float*)d_in[22], *b41 = (const float*)d_in[23],
              *s41 = (const float*)d_in[24], *t41 = (const float*)d_in[25];
  // netR_3
  const float *W30 = (const float*)d_in[26], *b30 = (const float*)d_in[27],
              *s30 = (const float*)d_in[28], *t30 = (const float*)d_in[29];
  const float *W31 = (const float*)d_in[30], *b31 = (const float*)d_in[31],
              *s31 = (const float*)d_in[32], *t31 = (const float*)d_in[33];
  const float *W32 = (const float*)d_in[34], *b32 = (const float*)d_in[35],
              *s32 = (const float*)d_in[36], *t32 = (const float*)d_in[37];
  // fc1 / fc2
  const float *Wf1 = (const float*)d_in[38], *bf1 = (const float*)d_in[39],
              *sf1 = (const float*)d_in[40], *tf1 = (const float*)d_in[41];
  const float *Wf2 = (const float*)d_in[42], *bf2 = (const float*)d_in[43];

  // ---- workspace carve (all chunks 16B aligned) ----
  char* ws = (char*)d_ws;
  float* xfeat = (float*)ws; ws += (size_t)32 * 512 * 128 * 4;   // (B,512,128)
  int*   idx1  = (int*)ws;   ws += (size_t)32 * 128 * 64 * 4;    // (B,128,64)
  float* h4    = (float*)ws; ws += (size_t)32 * 128 * 256 * 4;   // (B,128,256)
  int*   idx2  = (int*)ws;   ws += (size_t)32 * 32 * 64 * 4;     // (B,32,64)
  float* h2    = (float*)ws; ws += (size_t)32 * 32 * 256 * 4;    // (B,32,256)
  float* feat  = (float*)ws; ws += (size_t)32 * 1024 * 4;        // (B,1024) f32
  _Float16* featH = (_Float16*)ws; ws += (size_t)32 * 1024 * 2;  // (B,1024) f16
  // f16 padded weights
  _Float16* hW10 = (_Float16*)ws; ws += (size_t)64 * 32 * 2;
  _Float16* hW11 = (_Float16*)ws; ws += (size_t)128 * 64 * 2;
  _Float16* hW20 = (_Float16*)ws; ws += (size_t)128 * 160 * 2;
  _Float16* hW21 = (_Float16*)ws; ws += (size_t)256 * 128 * 2;
  _Float16* hW40 = (_Float16*)ws; ws += (size_t)128 * 160 * 2;
  _Float16* hW41 = (_Float16*)ws; ws += (size_t)256 * 128 * 2;
  _Float16* hW30 = (_Float16*)ws; ws += (size_t)256 * 288 * 2;
  _Float16* hW31 = (_Float16*)ws; ws += (size_t)512 * 256 * 2;
  _Float16* hW32 = (_Float16*)ws; ws += (size_t)1024 * 512 * 2;
  _Float16* hWf1 = (_Float16*)ws; ws += (size_t)512 * 1024 * 2;
  _Float16* hWf2 = (_Float16*)ws; ws += (size_t)96 * 512 * 2;

  // ---- weight prep (f32 -> padded f16) ----
#define WPREP(src, dst, co, K, KP)                                              \
  k_wprep<<<dim3((((co) * (KP)) + 255) / 256), dim3(256), 0, stream>>>(         \
      src, dst, co, K, KP, (co) * (KP))
  WPREP(W10, hW10, 64, 6, 32);
  WPREP(W11, hW11, 128, 64, 64);
  WPREP(W20, hW20, 128, 131, 160);
  WPREP(W21, hW21, 256, 128, 128);
  WPREP(W40, hW40, 128, 131, 160);
  WPREP(W41, hW41, 256, 128, 128);
  WPREP(W30, hW30, 256, 259, 288);
  WPREP(W31, hW31, 512, 256, 256);
  WPREP(W32, hW32, 1024, 512, 512);
  WPREP(Wf1, hWf1, 512, 1024, 1024);
  WPREP(Wf2, hWf2, 96, 512, 512);   // co padded 87 -> 96
#undef WPREP

  // ---- network ----
  k_net1<<<dim3(32 * 512), dim3(128), 0, stream>>>(x, hW10, b10, s10, t10,
                                                   hW11, b11, s11, t11, xfeat);
  k_knn<<<dim3(32 * 128), dim3(128), 0, stream>>>(y, 512, 128, 0, idx1);
  k_group_mlp<<<dim3(32 * 128), dim3(128), 0, stream>>>(y, xfeat, idx1, 128, 0,
      hW40, b40, s40, t40, hW41, b41, s41, t41, h4);                      // netR_4
  k_knn<<<dim3(32 * 32), dim3(128), 0, stream>>>(y, 256, 32, 1, idx2);
  k_group_mlp<<<dim3(32 * 32), dim3(128), 0, stream>>>(y, xfeat, idx2, 32, 1,
      hW20, b20, s20, t20, hW21, b21, s21, t21, h2);                      // netR_2
  k_zero<<<dim3((32 * 1024 + 255) / 256), dim3(256), 0, stream>>>(feat, 32 * 1024);
  k_net3<<<dim3(32 * 36), dim3(128), 0, stream>>>(y, h2, h4,
      hW30, b30, s30, t30, hW31, b31, s31, t31, hW32, b32, s32, t32, feat);
  k_tohalf<<<dim3((32 * 1024 + 255) / 256), dim3(256), 0, stream>>>(feat, featH, 32 * 1024);
  k_fc<<<dim3(1), dim3(128), 0, stream>>>(featH, hWf1, bf1, sf1, tf1,
                                          hWf2, bf2, (float*)d_out);
}
